// Actor_90529320665718
// MI455X (gfx1250) — matrix operand order repaired and checked by reading, not compiled, not run on hardware
//
#include <hip/hip_runtime.h>
#include <hip/hip_bf16.h>

typedef __attribute__((ext_vector_type(16))) _Float16 v16h;
typedef __attribute__((ext_vector_type(8)))  _Float16 v8h;
typedef __attribute__((ext_vector_type(8)))  float    v8f;

#define BATCH 16384
#define EPS 1e-5f

// ---------------------------------------------------------------------------
// f32 -> f16 conversion (grid-stride)
// ---------------------------------------------------------------------------
__global__ void cvt_f32_f16_kernel(const float* __restrict__ src,
                                   _Float16* __restrict__ dst, int n) {
  for (int i = blockIdx.x * blockDim.x + threadIdx.x; i < n;
       i += gridDim.x * blockDim.x)
    dst[i] = (_Float16)src[i];
}

// ---------------------------------------------------------------------------
// Pack row-major f16 weight [K, Nsrc] into per-lane WMMA B-fragment order.
// B fragment (f16 32x16): lane l: n = l&15, h = l>>4; element e (0..15):
//   value = W[k0 + (e>>3)*16 + h*8 + (e&7)][n0 + n]
// Tile (kT, nT) stored at ((nT * (K/32) + kT) * 512 + lane*16) halves.
// Columns >= Nsrc are zero-padded (used for the 4->16 padded promote head).
// ---------------------------------------------------------------------------
__global__ void pack_weight_kernel(const _Float16* __restrict__ W,
                                   _Float16* __restrict__ Wp,
                                   int K, int Nsrc) {
  const int k0 = blockIdx.x * 32;
  const int n0 = blockIdx.y * 16;
  const int lane = threadIdx.x;
  const int n = lane & 15, h = lane >> 4;
  union { v16h v; _Float16 e[16]; } u;
#pragma unroll
  for (int e = 0; e < 16; ++e) {
    int k = k0 + (e >> 3) * 16 + h * 8 + (e & 7);
    int nc = n0 + n;
    u.e[e] = (nc < Nsrc) ? W[(size_t)k * Nsrc + nc] : (_Float16)0.0f;
  }
  size_t tile = (size_t)blockIdx.y * gridDim.x + blockIdx.x;
  *reinterpret_cast<v16h*>(Wp + tile * 512 + (size_t)lane * 16) = u.v;
}

// ---------------------------------------------------------------------------
// BN fold: scale = g * rsqrt(v+eps); bias = be + (b - m) * scale
// ---------------------------------------------------------------------------
__global__ void bn_prep_kernel(const float* __restrict__ g,
                               const float* __restrict__ be,
                               const float* __restrict__ m,
                               const float* __restrict__ v,
                               const float* __restrict__ b,
                               float* __restrict__ scale,
                               float* __restrict__ bias, int n) {
  int i = blockIdx.x * blockDim.x + threadIdx.x;
  if (i < n) {
    float s = g[i] * rsqrtf(v[i] + EPS);
    scale[i] = s;
    bias[i] = be[i] + (b[i] - m[i]) * s;
  }
}

__global__ void pad_bias_kernel(const float* __restrict__ b,
                                float* __restrict__ out, int n, int npad) {
  int i = blockIdx.x * blockDim.x + threadIdx.x;
  if (i < npad) out[i] = (i < n) ? b[i] : 0.0f;
}

// ---------------------------------------------------------------------------
// Pack X = concat(embeddings[B,448], teams[B,64]) -> f16 [B,512]
// ---------------------------------------------------------------------------
__global__ void pack_x_kernel(const float* __restrict__ emb,
                              const float* __restrict__ teams,
                              _Float16* __restrict__ X, int total) {
  for (int i = blockIdx.x * blockDim.x + threadIdx.x; i < total;
       i += gridDim.x * blockDim.x) {
    int b = i >> 9, j = i & 511;
    float v = (j < 448) ? emb[(size_t)b * 448 + j] : teams[(size_t)b * 64 + (j - 448)];
    X[i] = (_Float16)v;
  }
}

// ---------------------------------------------------------------------------
// sel_filter[b,s] = any(move_matrices[b,s,:] != 0) as float {0,1}
// one block per batch row; 256 threads; 16B int4 loads; LDS OR-reduce
// ---------------------------------------------------------------------------
__global__ void sel_filter_kernel(const int* __restrict__ mv,
                                  float* __restrict__ mask) {
  __shared__ int red[256];
  const int t = threadIdx.x;
  const int s = t >> 2, q = t & 3;
  const int4* p = reinterpret_cast<const int4*>(
      mv + (size_t)blockIdx.x * 4096 + s * 64 + q * 16);
  int any = 0;
#pragma unroll
  for (int i = 0; i < 4; ++i) {
    int4 x = p[i];
    any |= (x.x | x.y | x.z | x.w);
  }
  red[t] = any;
  __syncthreads();
  if (q == 0) {
    int a = red[t] | red[t + 1] | red[t + 2] | red[t + 3];
    mask[(size_t)blockIdx.x * 64 + s] = a ? 1.0f : 0.0f;
  }
}

// ---------------------------------------------------------------------------
// WMMA GEMM: C[M,N] = A[M,K](f16,row-major) @ Bpacked + epilogue
//   - software-pipelined K loop: fragments double-buffered so WMMAs only
//     depend on loads issued one iteration earlier (partial loadcnt waits)
//   - optional per-column scale/bias (BN fold); nullptr scale => 1
//   - NEXTRA in {0,1,2}: fused one-hot rows: acc += Wext[off + sel[m]][n]
//   - RELU, output f16 (next layer A) or f32 (logits)
// Block = 256 threads = 8 waves; wave tile = 16(M) x 16*NFRAG(N).
// ---------------------------------------------------------------------------
template <int NFRAG, bool RELU, bool F16OUT, int NEXTRA>
__global__ __launch_bounds__(256) void gemm_wmma_kernel(
    const _Float16* __restrict__ A, const _Float16* __restrict__ Bp,
    const float* __restrict__ colScale, const float* __restrict__ colBias,
    _Float16* __restrict__ outH, float* __restrict__ outF,
    const _Float16* __restrict__ Wext, const int* __restrict__ sel0, int off0,
    const int* __restrict__ sel1, int off1, int N, int K) {
  const int lane = threadIdx.x & 31;
  const int wv = threadIdx.x >> 5;
  const int m0 = blockIdx.x * 128 + wv * 16;
  const int nT0 = blockIdx.y * NFRAG;
  const int r = lane & 15, h = lane >> 4;
  const int kTiles = K >> 5;

  v8f acc[NFRAG] = {};

  // A fragment (16x32 f16): lane (r,h); element e: K = (e>>3)*16 + h*8 + (e&7)
  const _Float16* aPtr = A + (size_t)(m0 + r) * K + h * 8;
  const _Float16* bPtr = Bp + (size_t)lane * 16;

  union AF { v16h v; v8h p[2]; };
  AF aCur, aNxt;
  v16h bCur[NFRAG], bNxt[NFRAG];

  // prologue: load tile 0
  aCur.p[0] = *reinterpret_cast<const v8h*>(aPtr);
  aCur.p[1] = *reinterpret_cast<const v8h*>(aPtr + 16);
#pragma unroll
  for (int f = 0; f < NFRAG; ++f)
    bCur[f] = *reinterpret_cast<const v16h*>(
        bPtr + (size_t)(nT0 + f) * kTiles * 512);

#pragma unroll 2
  for (int kt = 0; kt < kTiles - 1; ++kt) {
    // issue next tile's loads before consuming current tile
    const _Float16* aP = aPtr + (size_t)(kt + 1) * 32;
    aNxt.p[0] = *reinterpret_cast<const v8h*>(aP);
    aNxt.p[1] = *reinterpret_cast<const v8h*>(aP + 16);
#pragma unroll
    for (int f = 0; f < NFRAG; ++f)
      bNxt[f] = *reinterpret_cast<const v16h*>(
          bPtr + ((size_t)(nT0 + f) * kTiles + (kt + 1)) * 512);
#pragma unroll
    for (int f = 0; f < NFRAG; ++f)
      acc[f] = __builtin_amdgcn_wmma_f32_16x16x32_f16(
          false, aCur.v, false, bCur[f], (short)0, acc[f], false, false);
    aCur = aNxt;
#pragma unroll
    for (int f = 0; f < NFRAG; ++f) bCur[f] = bNxt[f];
  }
  // epilogue tile
#pragma unroll
  for (int f = 0; f < NFRAG; ++f)
    acc[f] = __builtin_amdgcn_wmma_f32_16x16x32_f16(
        false, aCur.v, false, bCur[f], (short)0, acc[f], false, false);

  // C/D layout (f32 16x16): VGPR v -> M = v + 8*h, N = lane&15
#pragma unroll
  for (int f = 0; f < NFRAG; ++f) {
    const int ncol = (nT0 + f) * 16 + r;
    const float sc = colScale ? colScale[ncol] : 1.0f;
    const float bi = colBias ? colBias[ncol] : 0.0f;
#pragma unroll
    for (int v = 0; v < 8; ++v) {
      const int mrow = m0 + v + 8 * h;
      float val = acc[f][v] * sc + bi;
      if constexpr (NEXTRA >= 1)
        val += (float)Wext[(size_t)(off0 + sel0[mrow]) * N + ncol];
      if constexpr (NEXTRA >= 2)
        val += (float)Wext[(size_t)(off1 + sel1[mrow]) * N + ncol];
      if constexpr (RELU) val = fmaxf(val, 0.0f);
      if constexpr (F16OUT)
        outH[(size_t)mrow * N + ncol] = (_Float16)val;
      else
        outF[(size_t)mrow * N + ncol] = val;
    }
  }
}

// ---------------------------------------------------------------------------
// Weighted masked softmax + argmax over 64 columns (one block per row).
// p_i = w_i * exp(x_i - xmax) / sum_j w_j * exp(x_j - xmax)
// ---------------------------------------------------------------------------
__global__ void select_head_kernel(const float* __restrict__ L,
                                   const float* __restrict__ MASK,
                                   int* __restrict__ sel,
                                   float* __restrict__ slogp) {
  __shared__ float sv[64];
  __shared__ int si[64];
  __shared__ float s_max, s_sum;
  const int b = blockIdx.x, t = threadIdx.x;
  float x = L[(size_t)b * 64 + t];
  float w = MASK[(size_t)b * 64 + t];
  sv[t] = x;
  __syncthreads();
  for (int st = 32; st > 0; st >>= 1) {
    if (t < st) sv[t] = fmaxf(sv[t], sv[t + st]);
    __syncthreads();
  }
  if (t == 0) s_max = sv[0];
  __syncthreads();
  float p = w * expf(x - s_max);
  sv[t] = p;
  __syncthreads();
  for (int st = 32; st > 0; st >>= 1) {
    if (t < st) sv[t] += sv[t + st];
    __syncthreads();
  }
  if (t == 0) s_sum = sv[0];
  __syncthreads();
  sv[t] = p;
  si[t] = t;
  __syncthreads();
  for (int st = 32; st > 0; st >>= 1) {
    if (t < st) {
      float ov = sv[t + st];
      int oi = si[t + st];
      if (ov > sv[t] || (ov == sv[t] && oi < si[t])) { sv[t] = ov; si[t] = oi; }
    }
    __syncthreads();
  }
  if (t == 0) {
    sel[b] = si[0];
    slogp[b] = logf(sv[0] / s_sum);
  }
}

__global__ void target_head_kernel(const float* __restrict__ L,
                                   const int* __restrict__ mvmat,
                                   const int* __restrict__ sel,
                                   int* __restrict__ tgt,
                                   float* __restrict__ tlogp,
                                   int* __restrict__ pf) {
  __shared__ float sv[64];
  __shared__ int si[64];
  __shared__ int smv[64];
  __shared__ float s_max, s_sum;
  const int b = blockIdx.x, t = threadIdx.x;
  const int s = sel[b];
  const int mv = mvmat[(size_t)b * 4096 + (size_t)s * 64 + t];
  float x = L[(size_t)b * 64 + t];
  float w = (float)mv;  // row values {0,1,2} weight the probabilities
  smv[t] = mv;
  sv[t] = x;
  __syncthreads();
  for (int st = 32; st > 0; st >>= 1) {
    if (t < st) sv[t] = fmaxf(sv[t], sv[t + st]);
    __syncthreads();
  }
  if (t == 0) s_max = sv[0];
  __syncthreads();
  float p = w * expf(x - s_max);
  sv[t] = p;
  __syncthreads();
  for (int st = 32; st > 0; st >>= 1) {
    if (t < st) sv[t] += sv[t + st];
    __syncthreads();
  }
  if (t == 0) s_sum = sv[0];
  __syncthreads();
  sv[t] = p;
  si[t] = t;
  __syncthreads();
  for (int st = 32; st > 0; st >>= 1) {
    if (t < st) {
      float ov = sv[t + st];
      int oi = si[t + st];
      if (ov > sv[t] || (ov == sv[t] && oi < si[t])) { sv[t] = ov; si[t] = oi; }
    }
    __syncthreads();
  }
  if (t == 0) {
    int tg = si[0];
    tgt[b] = tg;
    tlogp[b] = logf(sv[0] / s_sum);
    pf[b] = (smv[tg] == 2) ? 1 : 0;
  }
}

// ---------------------------------------------------------------------------
// Promote head (4-way softmax) + final output assembly.
// d_out layout (floats): [select(B) | target(B) | promote(B) | logp(B)]
// ---------------------------------------------------------------------------
__global__ void promote_final_kernel(const float* __restrict__ LP,
                                     const int* __restrict__ sel,
                                     const int* __restrict__ tgt,
                                     const int* __restrict__ pf,
                                     const float* __restrict__ slogp,
                                     const float* __restrict__ tlogp,
                                     float* __restrict__ out, int B) {
  for (int b = blockIdx.x * blockDim.x + threadIdx.x; b < B;
       b += gridDim.x * blockDim.x) {
    float l[4];
#pragma unroll
    for (int i = 0; i < 4; ++i) l[i] = LP[(size_t)b * 16 + i];
    float lmax = l[0];
    int arg = 0;
#pragma unroll
    for (int i = 1; i < 4; ++i)
      if (l[i] > lmax) { lmax = l[i]; arg = i; }
    float sum = 0.0f;
#pragma unroll
    for (int i = 0; i < 4; ++i) sum += expf(l[i] - lmax);
    // max(softmax) = 1/sum  ->  log(max pd) = -log(sum)
    int active = pf[b];
    float plogp = active ? -logf(sum) : 0.0f;
    int promote = active ? arg : -1;
    out[b] = (float)sel[b];
    out[B + b] = (float)tgt[b];
    out[2 * B + b] = (float)promote;
    out[3 * B + b] = slogp[b] + tlogp[b] + plogp;
  }
}

// ---------------------------------------------------------------------------
// Host orchestration
// ---------------------------------------------------------------------------
extern "C" void kernel_launch(void* const* d_in, const int* in_sizes, int n_in,
                              void* d_out, int out_size, void* d_ws,
                              size_t ws_size, hipStream_t stream) {
  (void)in_sizes; (void)n_in; (void)out_size; (void)ws_size;
  const int B = BATCH;

  // ---- inputs (setup_inputs dict order, params flattened in insert order)
  const float* emb   = (const float*)d_in[0];
  const float* teams = (const float*)d_in[1];
  const int*   mvm   = (const int*)d_in[2];
  const float* W1  = (const float*)d_in[3];
  const float* b1  = (const float*)d_in[4];
  const float* W2  = (const float*)d_in[5];
  const float* b2  = (const float*)d_in[6];
  const float* g1  = (const float*)d_in[7];
  const float* be1 = (const float*)d_in[8];
  const float* m1  = (const float*)d_in[9];
  const float* v1  = (const float*)d_in[10];
  const float* g2  = (const float*)d_in[11];
  const float* be2 = (const float*)d_in[12];
  const float* m2  = (const float*)d_in[13];
  const float* v2  = (const float*)d_in[14];
  const float* Ws1 = (const float*)d_in[15];
  const float* bs1 = (const float*)d_in[16];
  const float* Ws2 = (const float*)d_in[17];
  const float* bs2 = (const float*)d_in[18];
  const float* Wt1 = (const float*)d_in[19];
  const float* bt1 = (const float*)d_in[20];
  const float* Wt2 = (const float*)d_in[21];
  const float* bt2 = (const float*)d_in[22];
  const float* Wp1 = (const float*)d_in[23];
  const float* bp1 = (const float*)d_in[24];
  const float* Wp2 = (const float*)d_in[25];
  const float* bp2 = (const float*)d_in[26];

  // ---- workspace bump allocator (256B aligned)
  char* ws = (char*)d_ws;
  size_t off = 0;
  auto alloc = [&](size_t bytes) -> char* {
    char* p = ws + off;
    off = (off + bytes + 255) & ~(size_t)255;
    return p;
  };
  // row-major f16 weights
  _Float16* W1h  = (_Float16*)alloc(512 * 512 * 2);
  _Float16* W2h  = (_Float16*)alloc(512 * 512 * 2);
  _Float16* Ws1h = (_Float16*)alloc(512 * 512 * 2);
  _Float16* Ws2h = (_Float16*)alloc(512 * 64 * 2);
  _Float16* Wt1h = (_Float16*)alloc(576 * 512 * 2);
  _Float16* Wt2h = (_Float16*)alloc(512 * 64 * 2);
  _Float16* Wp1h = (_Float16*)alloc(640 * 512 * 2);
  _Float16* Wp2h = (_Float16*)alloc(512 * 4 * 2);
  // packed B fragments (K=512 portion only)
  _Float16* W1p  = (_Float16*)alloc(512 * 512 * 2);
  _Float16* W2p  = (_Float16*)alloc(512 * 512 * 2);
  _Float16* Ws1p = (_Float16*)alloc(512 * 512 * 2);
  _Float16* Ws2p = (_Float16*)alloc(512 * 64 * 2);
  _Float16* Wt1p = (_Float16*)alloc(512 * 512 * 2);
  _Float16* Wt2p = (_Float16*)alloc(512 * 64 * 2);
  _Float16* Wp1p = (_Float16*)alloc(512 * 512 * 2);
  _Float16* Wp2p = (_Float16*)alloc(512 * 16 * 2);  // N padded 4->16
  // folded BN scale/bias + padded promote bias
  float* s1c = (float*)alloc(512 * 4);
  float* b1c = (float*)alloc(512 * 4);
  float* s2c = (float*)alloc(512 * 4);
  float* b2c = (float*)alloc(512 * 4);
  float* bp2pad = (float*)alloc(16 * 4);
  // activations
  _Float16* X16  = (_Float16*)alloc((size_t)B * 512 * 2);
  _Float16* H16  = (_Float16*)alloc((size_t)B * 512 * 2);
  _Float16* P16  = (_Float16*)alloc((size_t)B * 512 * 2);
  _Float16* HB   = (_Float16*)alloc((size_t)B * 512 * 2);
  float* L64  = (float*)alloc((size_t)B * 64 * 4);
  float* MASK = (float*)alloc((size_t)B * 64 * 4);
  float* LP   = (float*)alloc((size_t)B * 16 * 4);
  int*   SEL  = (int*)alloc((size_t)B * 4);
  int*   TGT  = (int*)alloc((size_t)B * 4);
  int*   PF   = (int*)alloc((size_t)B * 4);
  float* SLP  = (float*)alloc((size_t)B * 4);
  float* TLP  = (float*)alloc((size_t)B * 4);

  // ---- weight prep
  auto cvt = [&](const float* s, _Float16* d, int n) {
    cvt_f32_f16_kernel<<<(n + 255) / 256, 256, 0, stream>>>(s, d, n);
  };
  cvt(W1, W1h, 512 * 512);  cvt(W2, W2h, 512 * 512);
  cvt(Ws1, Ws1h, 512 * 512); cvt(Ws2, Ws2h, 512 * 64);
  cvt(Wt1, Wt1h, 576 * 512); cvt(Wt2, Wt2h, 512 * 64);
  cvt(Wp1, Wp1h, 640 * 512); cvt(Wp2, Wp2h, 512 * 4);

  bn_prep_kernel<<<2, 256, 0, stream>>>(g1, be1, m1, v1, b1, s1c, b1c, 512);
  bn_prep_kernel<<<2, 256, 0, stream>>>(g2, be2, m2, v2, b2, s2c, b2c, 512);
  pad_bias_kernel<<<1, 32, 0, stream>>>(bp2, bp2pad, 4, 16);

  pack_weight_kernel<<<dim3(16, 32), 32, 0, stream>>>(W1h, W1p, 512, 512);
  pack_weight_kernel<<<dim3(16, 32), 32, 0, stream>>>(W2h, W2p, 512, 512);
  pack_weight_kernel<<<dim3(16, 32), 32, 0, stream>>>(Ws1h, Ws1p, 512, 512);
  pack_weight_kernel<<<dim3(16, 4), 32, 0, stream>>>(Ws2h, Ws2p, 512, 64);
  pack_weight_kernel<<<dim3(16, 32), 32, 0, stream>>>(Wt1h, Wt1p, 512, 512);
  pack_weight_kernel<<<dim3(16, 4), 32, 0, stream>>>(Wt2h, Wt2p, 512, 64);
  pack_weight_kernel<<<dim3(16, 32), 32, 0, stream>>>(Wp1h, Wp1p, 512, 512);
  pack_weight_kernel<<<dim3(16, 1), 32, 0, stream>>>(Wp2h, Wp2p, 512, 4);

  // ---- input prep (concat -> f16) and sel_filter scan (256 MB, BW-bound)
  pack_x_kernel<<<4096, 256, 0, stream>>>(emb, teams, X16, B * 512);
  sel_filter_kernel<<<B, 256, 0, stream>>>(mvm, MASK);

  const dim3 g512(B / 128, 8), g64(B / 128, 1), blk(256);

  // ---- trunk: GEMM + BN + ReLU
  gemm_wmma_kernel<4, true, true, 0><<<g512, blk, 0, stream>>>(
      X16, W1p, s1c, b1c, H16, nullptr, nullptr, nullptr, 0, nullptr, 0, 512, 512);
  gemm_wmma_kernel<4, true, true, 0><<<g512, blk, 0, stream>>>(
      H16, W2p, s2c, b2c, P16, nullptr, nullptr, nullptr, 0, nullptr, 0, 512, 512);

  // ---- select head
  gemm_wmma_kernel<4, true, true, 0><<<g512, blk, 0, stream>>>(
      P16, Ws1p, nullptr, bs1, HB, nullptr, nullptr, nullptr, 0, nullptr, 0, 512, 512);
  gemm_wmma_kernel<4, false, false, 0><<<g64, blk, 0, stream>>>(
      HB, Ws2p, nullptr, bs2, nullptr, L64, nullptr, nullptr, 0, nullptr, 0, 64, 512);
  select_head_kernel<<<B, 64, 0, stream>>>(L64, MASK, SEL, SLP);

  // ---- target head (one-hot fused as +Wt1[512+select] row)
  gemm_wmma_kernel<4, true, true, 1><<<g512, blk, 0, stream>>>(
      P16, Wt1p, nullptr, bt1, HB, nullptr, Wt1h, SEL, 512, nullptr, 0, 512, 512);
  gemm_wmma_kernel<4, false, false, 0><<<g64, blk, 0, stream>>>(
      HB, Wt2p, nullptr, bt2, nullptr, L64, nullptr, nullptr, 0, nullptr, 0, 64, 512);
  target_head_kernel<<<B, 64, 0, stream>>>(L64, mvm, SEL, TGT, TLP, PF);

  // ---- promote head (two one-hot rows fused)
  gemm_wmma_kernel<4, true, true, 2><<<g512, blk, 0, stream>>>(
      P16, Wp1p, nullptr, bp1, HB, nullptr, Wp1h, SEL, 512, TGT, 576, 512, 512);
  gemm_wmma_kernel<1, false, false, 0><<<g64, blk, 0, stream>>>(
      HB, Wp2p, nullptr, bp2pad, nullptr, LP, nullptr, nullptr, 0, nullptr, 0, 16, 512);

  promote_final_kernel<<<(B + 255) / 256, 256, 0, stream>>>(
      LP, SEL, TGT, PF, SLP, TLP, (float*)d_out, B);
}
